// CoreDotProductAttention_10977936409095
// MI455X (gfx1250) — matrix operand order repaired and checked by reading, not compiled, not run on hardware
//
#include <hip/hip_runtime.h>
#include <hip/hip_bf16.h>

typedef _Float16 v16h __attribute__((ext_vector_type(16)));
typedef _Float16 v8h  __attribute__((ext_vector_type(8)));
typedef float    v8f  __attribute__((ext_vector_type(8)));

#define WMMA_F16(A, B, C) \
  __builtin_amdgcn_wmma_f32_16x16x32_f16(false, (A), false, (B), (short)0, (C), false, false)

constexpr int BS = 2, SEQ = 2048, EMBED = 1024, HEADS = 16, HD = 64;
constexpr int WAVES = 8;           // waves per 256-thread block (wave32)

// ---------------------------------------------------------------------------
// A-operand (16 x K=32, f16) gather from f16 memory.
// Layout: lanes l and l+16 hold row M=l&15; halves 0..15 = K ko..ko+15,
// ko = (lane<16) ? 0 : 16 (contiguous K order).  Two 16-byte loads.
// ---------------------------------------------------------------------------
__device__ inline v16h loadA_f16(const _Float16* __restrict__ p) {
  v8h a0 = *(const v8h*)p;
  v8h a1 = *(const v8h*)(p + 8);
  v16h a;
  #pragma unroll
  for (int i = 0; i < 8; ++i) { a[i] = a0[i]; a[8 + i] = a1[i]; }
  return a;
}

// ---------------------------------------------------------------------------
// Kernel 1: flash attention, one wave per 32-query strip of one (b,h):
// two 16-query sets share every K/V A-tile (2x WMMA per byte loaded).
// S^T = K*Q^T (query in lanes), O^T = V^T * P^T, online softmax.
// K operand from Kh (f16, [b,s,E]); V operand from Vt (f16, [bh*64+d, s]).
// Writes attention output X as f16 [BS*SEQ, EMBED].
// ---------------------------------------------------------------------------
__global__ __launch_bounds__(256)
void attn_kernel(const float* __restrict__ Q, const _Float16* __restrict__ Kh,
                 const _Float16* __restrict__ Vt, const int* __restrict__ mask,
                 _Float16* __restrict__ X) {
  const int lane = threadIdx.x & 31;
  const int wave = threadIdx.x >> 5;
  const int tile = blockIdx.x * WAVES + wave;
  const int QSTRIPS = SEQ / 32;             // 64 strips per (b,h)
  const int qt = tile % QSTRIPS;
  const int bh = tile / QSTRIPS;
  const int h  = bh % HEADS;
  const int b  = bh / HEADS;

  const int  n  = lane & 15;      // query column owned by this lane
  const bool hi = lane >= 16;
  const int  m  = lane & 15;      // A-operand row
  const int  ko = hi ? 16 : 0;    // A-operand K offset

  const float*    Qp  = Q  + ((size_t)b * SEQ) * EMBED + (size_t)h * HD;
  const _Float16* Kp  = Kh + ((size_t)b * SEQ) * EMBED + (size_t)h * HD;
  const _Float16* Vtp = Vt + (size_t)bh * HD * SEQ;

  int qrow[2];
  qrow[0] = qt * 32 + n;
  qrow[1] = qt * 32 + 16 + n;
  const int* mrow[2];
  mrow[0] = mask + ((size_t)b * SEQ + qrow[0]) * SEQ;
  mrow[1] = mask + ((size_t)b * SEQ + qrow[1]) * SEQ;

  // Q tiles once, in B-operand layout: halves 0..15 = d = 32c + (hi?16:0) + 0..15
  v16h qB[2][2];
  #pragma unroll
  for (int qs = 0; qs < 2; ++qs) {
    const float* qr = Qp + (size_t)qrow[qs] * EMBED + (hi ? 16 : 0);
    #pragma unroll
    for (int c = 0; c < 2; ++c) {
      const float4* qv = (const float4*)(qr + 32 * c);
      #pragma unroll
      for (int i = 0; i < 4; ++i) {
        float4 f = qv[i];
        qB[qs][c][4*i+0] = (_Float16)f.x; qB[qs][c][4*i+1] = (_Float16)f.y;
        qB[qs][c][4*i+2] = (_Float16)f.z; qB[qs][c][4*i+3] = (_Float16)f.w;
      }
    }
  }

  v8f acc[2][4] = {};                    // O^T accumulators per query set
  float m_run[2] = {-1e30f, -1e30f};
  float l_run[2] = {0.0f, 0.0f};
  const float scale = 0.125f;            // 1/sqrt(64)

  for (int kb = 0; kb < SEQ; kb += 32) {
    // ---- K A-tiles, shared by both query sets
    const _Float16* ka = Kp + (size_t)(kb + m) * EMBED + ko;
    v16h kA00 = loadA_f16(ka);                          // rows kb,    d 0..31
    v16h kA01 = loadA_f16(ka + 32);                     // rows kb,    d 32..63
    v16h kA10 = loadA_f16(ka + (size_t)16 * EMBED);     // rows kb+16, d 0..31
    v16h kA11 = loadA_f16(ka + (size_t)16 * EMBED + 32);

    v8f s[2][2];
    #pragma unroll
    for (int qs = 0; qs < 2; ++qs) {
      v8f z0 = {}; v8f z1 = {};
      z0 = WMMA_F16(kA00, qB[qs][0], z0);
      z0 = WMMA_F16(kA01, qB[qs][1], z0);
      z1 = WMMA_F16(kA10, qB[qs][0], z1);
      z1 = WMMA_F16(kA11, qB[qs][1], z1);
      s[qs][0] = z0; s[qs][1] = z1;
    }

    // ---- V A-tiles, shared by both query sets
    const _Float16* va = Vtp + (size_t)m * SEQ + kb + ko;
    v16h vA0 = loadA_f16(va);
    v16h vA1 = loadA_f16(va + (size_t)16 * SEQ);
    v16h vA2 = loadA_f16(va + (size_t)32 * SEQ);
    v16h vA3 = loadA_f16(va + (size_t)48 * SEQ);

    v16h pB[2];
    #pragma unroll
    for (int qs = 0; qs < 2; ++qs) {
      // ---- mask + scale. Lane owns keys mk+r (tile0) and mk+16+r (tile1).
      const int mk = kb + (hi ? 8 : 0);
      const int* mr = mrow[qs];
      int4 mm0 = *(const int4*)(mr + mk);
      int4 mm1 = *(const int4*)(mr + mk + 4);
      int4 mm2 = *(const int4*)(mr + mk + 16);
      int4 mm3 = *(const int4*)(mr + mk + 20);
      const int mv[16] = {mm0.x, mm0.y, mm0.z, mm0.w, mm1.x, mm1.y, mm1.z, mm1.w,
                          mm2.x, mm2.y, mm2.z, mm2.w, mm3.x, mm3.y, mm3.z, mm3.w};
      float sc[16];
      #pragma unroll
      for (int r = 0; r < 8; ++r) {
        sc[r]     = (mv[r]     == 0) ? -1e9f : s[qs][0][r] * scale;
        sc[8 + r] = (mv[8 + r] == 0) ? -1e9f : s[qs][1][r] * scale;
      }

      // ---- online softmax (per-query = per lane-pair, shfl_xor 16 combines)
      float mx = sc[0];
      #pragma unroll
      for (int t = 1; t < 16; ++t) mx = fmaxf(mx, sc[t]);
      mx = fmaxf(mx, __shfl_xor(mx, 16));
      const float m_new = fmaxf(m_run[qs], mx);
      const float alpha = __expf(m_run[qs] - m_new);
      float p0[8], p1[8], ssum = 0.0f;
      #pragma unroll
      for (int r = 0; r < 8; ++r) {
        p0[r] = __expf(sc[r]     - m_new);
        p1[r] = __expf(sc[8 + r] - m_new);
        ssum += p0[r] + p1[r];
      }
      ssum += __shfl_xor(ssum, 16);
      l_run[qs] = l_run[qs] * alpha + ssum;
      m_run[qs] = m_new;

      #pragma unroll
      for (int t = 0; t < 4; ++t)
        #pragma unroll
        for (int r = 0; r < 8; ++r) acc[qs][t][r] *= alpha;

      // ---- rearrange P (score D-layout) into P^T B-operand layout:
      // lanes<16 take keys 0..15 (own tile0 + partner's), lanes>=16 keys 16..31.
      #pragma unroll
      for (int r = 0; r < 8; ++r) {
        const float o0 = __shfl_xor(p0[r], 16);
        const float o1 = __shfl_xor(p1[r], 16);
        pB[qs][r]     = (_Float16)(hi ? o1    : p0[r]);
        pB[qs][8 + r] = (_Float16)(hi ? p1[r] : o0);
      }
    }

    // ---- O^T += V^T * P^T : each V tile feeds both query sets
    #pragma unroll
    for (int qs = 0; qs < 2; ++qs) {
      acc[qs][0] = WMMA_F16(vA0, pB[qs], acc[qs][0]);
      acc[qs][1] = WMMA_F16(vA1, pB[qs], acc[qs][1]);
      acc[qs][2] = WMMA_F16(vA2, pB[qs], acc[qs][2]);
      acc[qs][3] = WMMA_F16(vA3, pB[qs], acc[qs][3]);
    }

    if (kb + 32 < SEQ) {
      __builtin_prefetch(Kp + (size_t)(kb + 32) * EMBED + lane * 16, 0, 1);
      __builtin_prefetch(Vtp + (size_t)(lane * 2) * SEQ + kb + 32, 0, 1);
    }
  }

  // ---- normalize and store X (f16): element (d = 16t + r + hi*8, q = qrow)
  #pragma unroll
  for (int qs = 0; qs < 2; ++qs) {
    const float inv = 1.0f / l_run[qs];
    _Float16* xr = X + ((size_t)b * SEQ + qrow[qs]) * EMBED + (size_t)h * HD + (hi ? 8 : 0);
    #pragma unroll
    for (int t = 0; t < 4; ++t)
      #pragma unroll
      for (int r = 0; r < 8; ++r)
        xr[t * 16 + r] = (_Float16)(acc[qs][t][r] * inv);
  }
}

// ---------------------------------------------------------------------------
// Kernel 2: elementwise f32 -> f16 convert (used for W and K)
// ---------------------------------------------------------------------------
__global__ __launch_bounds__(256)
void cvt_f32_to_f16(const float* __restrict__ src, _Float16* __restrict__ dst, int n) {
  int i = (blockIdx.x * blockDim.x + threadIdx.x) * 4;
  const int stride = gridDim.x * blockDim.x * 4;
  for (; i < n; i += stride) {
    float4 f = *(const float4*)(src + i);
    _Float16* d = dst + i;
    d[0] = (_Float16)f.x; d[1] = (_Float16)f.y;
    d[2] = (_Float16)f.z; d[3] = (_Float16)f.w;
  }
}

// ---------------------------------------------------------------------------
// Kernel 3: per-(b,h) transpose+convert: Vt[bh*64 + d, s] = (f16) V[b, s, h*64+d]
// LDS-tiled 32x32 so both global accesses are coalesced.
// grid: (SEQ/32, BS*HEADS*2), 256 threads (32 cols x 8 row-groups of 4)
// ---------------------------------------------------------------------------
__global__ __launch_bounds__(256)
void transpose_v(const float* __restrict__ V, _Float16* __restrict__ Vt) {
  __shared__ _Float16 t[32][33];
  const int s0 = blockIdx.x * 32;
  const int dt = blockIdx.y & 1;          // which 32-wide d tile
  const int bh = blockIdx.y >> 1;
  const int h = bh % HEADS, b = bh / HEADS;
  const int j  = threadIdx.x & 31;        // inner (d on load, s on store)
  const int g  = threadIdx.x >> 5;        // row group 0..7

  const float* src = V + ((size_t)b * SEQ + s0) * EMBED + h * HD + dt * 32 + j;
  #pragma unroll
  for (int r = 0; r < 4; ++r) {
    const int i = g * 4 + r;              // s offset
    t[i][j] = (_Float16)src[(size_t)i * EMBED];
  }
  __syncthreads();
  _Float16* dst = Vt + ((size_t)bh * HD + dt * 32) * SEQ + s0;
  #pragma unroll
  for (int r = 0; r < 4; ++r) {
    const int d = g * 4 + r;
    dst[(size_t)d * SEQ + j] = t[j][d];
  }
}

// ---------------------------------------------------------------------------
// Kernel 4: Y = X @ W^T + b. Wave computes a 32x64 tile: two 16-row A tiles
// share every W B-tile (2x WMMA per W byte). W rows are directly the
// contiguous-K B-operand layout (contraction over W's fast axis).
// ---------------------------------------------------------------------------
__global__ __launch_bounds__(256)
void proj_kernel(const _Float16* __restrict__ X, const _Float16* __restrict__ Wh,
                 const float* __restrict__ bias, float* __restrict__ Y) {
  const int lane = threadIdx.x & 31;
  const int wave = threadIdx.x >> 5;
  const int tile = blockIdx.x * WAVES + wave;
  const int NT = EMBED / 64;                 // 16 column tiles
  const int ntile = tile % NT;
  const int mtile = tile / NT;               // 32-row strips

  const int  n  = lane & 15;
  const bool hi = lane >= 16;

  const _Float16* xr0 = X + (size_t)(mtile * 32 + n)      * EMBED + (hi ? 16 : 0);
  const _Float16* xr1 = X + (size_t)(mtile * 32 + 16 + n) * EMBED + (hi ? 16 : 0);
  const _Float16* wr[4];
  #pragma unroll
  for (int s = 0; s < 4; ++s)
    wr[s] = Wh + (size_t)(ntile * 64 + s * 16 + n) * EMBED + (hi ? 16 : 0);

  v8f acc[2][4] = {};
  for (int k0 = 0; k0 < EMBED; k0 += 32) {
    v16h A0 = loadA_f16(xr0 + k0);
    v16h A1 = loadA_f16(xr1 + k0);
    #pragma unroll
    for (int s = 0; s < 4; ++s) {
      v16h Bv = *(const v16h*)(wr[s] + k0);
      acc[0][s] = WMMA_F16(A0, Bv, acc[0][s]);
      acc[1][s] = WMMA_F16(A1, Bv, acc[1][s]);
    }
  }

  #pragma unroll
  for (int g = 0; g < 2; ++g) {
    #pragma unroll
    for (int s = 0; s < 4; ++s) {
      const int j = ntile * 64 + s * 16 + n;
      const float bj = bias[j];
      #pragma unroll
      for (int r = 0; r < 8; ++r) {
        const int mm = mtile * 32 + g * 16 + r + (hi ? 8 : 0);
        Y[(size_t)mm * EMBED + j] = acc[g][s][r] + bj;
      }
    }
  }
}

// ---------------------------------------------------------------------------
extern "C" void kernel_launch(void* const* d_in, const int* in_sizes, int n_in,
                              void* d_out, int out_size, void* d_ws, size_t ws_size,
                              hipStream_t stream) {
  const float* q    = (const float*)d_in[0];
  const float* k    = (const float*)d_in[1];
  const float* v    = (const float*)d_in[2];
  const int*   mask = (const int*)d_in[3];
  const float* W    = (const float*)d_in[4];
  const float* bias = (const float*)d_in[5];
  float* out = (float*)d_out;

  const size_t NTOK = (size_t)BS * SEQ;            // 4096 tokens
  char* ws = (char*)d_ws;
  _Float16* X  = (_Float16*)ws;                    ws += NTOK * EMBED * 2;          // 8 MB
  _Float16* Wh = (_Float16*)ws;                    ws += (size_t)EMBED * EMBED * 2; // 2 MB
  _Float16* Kh = (_Float16*)ws;                    ws += NTOK * EMBED * 2;          // 8 MB
  _Float16* Vt = (_Float16*)ws;                                                     // 8 MB

  cvt_f32_to_f16<<<512, 256, 0, stream>>>(W, Wh, EMBED * EMBED);
  cvt_f32_to_f16<<<1024, 256, 0, stream>>>(k, Kh, (int)(NTOK * EMBED));
  transpose_v<<<dim3(SEQ / 32, BS * HEADS * 2), 256, 0, stream>>>(v, Vt);

  const int attn_blocks = BS * HEADS * (SEQ / 32) / WAVES;         // 256
  attn_kernel<<<attn_blocks, 256, 0, stream>>>(q, Kh, Vt, mask, X);

  const int proj_blocks = ((int)NTOK / 32) * (EMBED / 64) / WAVES; // 256
  proj_kernel<<<proj_blocks, 256, 0, stream>>>(X, Wh, bias, out);
}
